// MultiHeadSelfAttention_9216999817673
// MI455X (gfx1250) — matrix operand order repaired and checked by reading, not compiled, not run on hardware
//
#include <hip/hip_runtime.h>

typedef __attribute__((ext_vector_type(16))) _Float16 v16h;
typedef __attribute__((ext_vector_type(8)))  _Float16 v8h;
typedef __attribute__((ext_vector_type(2)))  __fp16   v2fp;
typedef __attribute__((ext_vector_type(8)))  float    v8f;

#define DEVINL __device__ __forceinline__

// ---------------------------------------------------------------------------
// WMMA fragment loaders (wave32); K is contiguous per lane half, not the ISA table order.
// A-matrix 16x32 f16: lane<16 -> row M=lane, K=k0..k0+15 (contiguous)
//                     lane>=16 -> row M=lane-16, K=k0+16..k0+31 (contiguous)
// Source matrix is row-major with K contiguous (leading dim = ldk elements).
DEVINL v16h load_frag_a(const _Float16* base, int ldk, int m0, int k0, int lane) {
  int l16 = lane & 15, hi = lane >> 4;
  const _Float16* p = base + (size_t)(m0 + l16) * (size_t)ldk + (k0 + hi * 16);
  v8h x0 = *(const v8h*)p;
  v8h x1 = *(const v8h*)(p + 8);
  v16h r;
#pragma unroll
  for (int i = 0; i < 8; ++i) { r[i] = x0[i]; r[i + 8] = x1[i]; }
  return r;
}

// B-matrix 32x16 f16 (KxN): lane holds column N = n0 + lane%16,
// K = k0 + 16*(lane/16) + 0..15 (contiguous). Source rows are the math-B
// columns, K contiguous (i.e. row n of "base" is column n of B).
DEVINL v16h load_frag_b(const _Float16* base, int ldk, int n0, int k0, int lane) {
  const _Float16* p = base + (size_t)(n0 + (lane & 15)) * (size_t)ldk
                           + (k0 + (lane >> 4) * 16);
  v8h x0 = *(const v8h*)p;
  v8h x1 = *(const v8h*)(p + 8);
  v16h r;
#pragma unroll
  for (int i = 0; i < 8; ++i) { r[i] = x0[i]; r[i + 8] = x1[i]; }
  return r;
}

DEVINL v8f wmma_f16(v16h a, v16h b, v8f c) {
  return __builtin_amdgcn_wmma_f32_16x16x32_f16(false, a, false, b,
                                                (short)0, c, false, false);
}

DEVINL unsigned pack2h(float a, float b) {  // v_cvt_pkrtz_f16_f32
  union { v2fp h; unsigned u; } c;
  c.h = __builtin_amdgcn_cvt_pkrtz(a, b);
  return c.u;
}

// ---------------------------------------------------------------------------
// fp32 -> f16 convert
__global__ void f32_to_f16_kernel(const float* __restrict__ in,
                                  _Float16* __restrict__ out, int n) {
  int i = blockIdx.x * blockDim.x + threadIdx.x;
  if (i < n) out[i] = (_Float16)in[i];
}

// ---------------------------------------------------------------------------
// C[M,N] = A[M,K] * B[N,K]^T ; 4 waves/block, 64x64 tile per wave (4x4 WMMA).
template <int KDIM, bool F16OUT>
__global__ __launch_bounds__(128) void gemm_wmma_kernel(
    const _Float16* __restrict__ A, const _Float16* __restrict__ B,
    void* __restrict__ Cv, int ldc) {
  int lane = threadIdx.x & 31;
  int wave = threadIdx.x >> 5;
  int m0 = blockIdx.x * 128 + (wave >> 1) * 64;
  int n0 = blockIdx.y * 128 + (wave & 1) * 64;

  v8f z = {};
  v8f acc[4][4];
#pragma unroll
  for (int i = 0; i < 4; ++i)
#pragma unroll
    for (int j = 0; j < 4; ++j) acc[i][j] = z;

  for (int k = 0; k < KDIM; k += 32) {
    v16h af[4], bf[4];
#pragma unroll
    for (int mi = 0; mi < 4; ++mi) af[mi] = load_frag_a(A, KDIM, m0 + mi * 16, k, lane);
#pragma unroll
    for (int ni = 0; ni < 4; ++ni) bf[ni] = load_frag_b(B, KDIM, n0 + ni * 16, k, lane);
    if (k + 64 < KDIM) {  // global_prefetch_b8 of next K slab
      __builtin_prefetch(A + (size_t)(m0 + (lane & 15)) * KDIM + k + 64, 0, 1);
      __builtin_prefetch(B + (size_t)(n0 + (lane & 15)) * KDIM + k + 64, 0, 1);
    }
#pragma unroll
    for (int mi = 0; mi < 4; ++mi)
#pragma unroll
      for (int ni = 0; ni < 4; ++ni)
        acc[mi][ni] = wmma_f16(af[mi], bf[ni], acc[mi][ni]);
  }

  int l16 = lane & 15, hi = lane >> 4;
#pragma unroll
  for (int mi = 0; mi < 4; ++mi)
#pragma unroll
    for (int ni = 0; ni < 4; ++ni)
#pragma unroll
      for (int r = 0; r < 8; ++r) {
        int row = m0 + mi * 16 + r + 8 * hi;   // C layout: VGPR r, lane half
        int col = n0 + ni * 16 + l16;
        if constexpr (F16OUT)
          ((_Float16*)Cv)[(size_t)row * ldc + col] = (_Float16)acc[mi][ni][r];
        else
          ((float*)Cv)[(size_t)row * ldc + col] = acc[mi][ni][r];
      }
}

// ---------------------------------------------------------------------------
// RoPE in-place on q and k sections of qkv (f16, layout [b, s, 3*1024]).
// The attention scale 1/sqrt(64) = 0.125 is folded into q here.
__global__ void rope_kernel(_Float16* __restrict__ qkv,
                            const int* __restrict__ pos) {
  const int S = 2048;
  int idx = blockIdx.x * blockDim.x + threadIdx.x;  // 2 * S * 16 * 32 threads
  int i = idx & 31;
  int h = (idx >> 5) & 15;
  int s = (idx >> 9) & (S - 1);
  int b = idx >> 20;
  if (b >= 2) return;
  float p = (float)pos[s];
  // inv_freq = 10000^(-2i/64) = exp(-2i/64 * ln(10000))
  float freq = __expf(-(float)(2 * i) * (9.210340371976184f / 64.0f));
  float ang = p * freq;
  float sn, c;
  __sincosf(ang, &sn, &c);
  size_t base = ((size_t)(b * S + s)) * 3072 + h * 64 + 2 * i;
  float qe = (float)qkv[base], qo = (float)qkv[base + 1];
  qkv[base]     = (_Float16)((qe * c - qo * sn) * 0.125f);
  qkv[base + 1] = (_Float16)((qe * sn + qo * c) * 0.125f);
  float ke = (float)qkv[base + 1024], ko = (float)qkv[base + 1025];
  qkv[base + 1024] = (_Float16)(ke * c - ko * sn);
  qkv[base + 1025] = (_Float16)(ke * sn + ko * c);
}

// ---------------------------------------------------------------------------
// One-time V transpose: qkv v-section [b, s, h*64+d] -> vt[(b*16+h)][d][s].
// LDS-tiled so both global sides are coalesced.
__global__ __launch_bounds__(256) void transpose_v_kernel(
    const _Float16* __restrict__ qkv, _Float16* __restrict__ vt) {
  const int S = 2048, TD = 3072;
  __shared__ _Float16 tile[64][17];          // [s-in-tile][d-in-tile], padded
  int bh = blockIdx.y;                       // b*16 + h
  int b = bh >> 4, h = bh & 15;
  int s0 = blockIdx.x * 64;
  int d0 = blockIdx.z * 16;
  const _Float16* src = qkv + (size_t)b * S * TD + 2048 + h * 64;
  _Float16* dst = vt + (size_t)bh * 64 * S;

#pragma unroll
  for (int pass = 0; pass < 4; ++pass) {
    int t = threadIdx.x;
    int d = t & 15;
    int sl = (t >> 4) + pass * 16;
    tile[sl][d] = src[(size_t)(s0 + sl) * TD + d0 + d];
  }
  __syncthreads();
#pragma unroll
  for (int pass = 0; pass < 4; ++pass) {
    int t = threadIdx.x;
    int sl = t & 63;
    int d = (t >> 6) + pass * 4;
    dst[(size_t)(d0 + d) * S + s0 + sl] = tile[sl][d];
  }
}

// ---------------------------------------------------------------------------
// Flash attention, transposed-score form. Block: 128 q-rows of one (b,h);
// 4 independent waves x 32 rows, no LDS, no barriers.
//   S^T(kv x q)  = K * Q^T          (A = K rows, B = Q^T columns, both direct)
//   O^T(d x q)  += V^T * P^T        (A = V^T rows from pre-transposed V)
// In the C layout of S^T a q-row lives in lanes {n, n+16}, so softmax stats
// are per-lane scalars with a single shfl_xor(16) combine, and the P^T
// B-fragment is assembled in-register with one packed-f16 xor-16 exchange.
__global__ __launch_bounds__(128) void attn_kernel(
    const _Float16* __restrict__ qkv, const _Float16* __restrict__ vt,
    _Float16* __restrict__ att) {
  const int S = 2048, TD = 3072;
  int qb = blockIdx.x & 15;          // S/128 = 16 q blocks
  int h  = (blockIdx.x >> 4) & 15;
  int b  = blockIdx.x >> 8;
  int lane = threadIdx.x & 31;
  int wave = threadIdx.x >> 5;
  int l16 = lane & 15, hi = lane >> 4;

  const _Float16* qbase = qkv + (size_t)b * S * TD + h * 64;
  const _Float16* kbase = qbase + 1024;
  const _Float16* vtb = vt + (size_t)((b << 4) + h) * 64 * S;  // [d][s]

  int m0 = qb * 128 + wave * 32;         // this wave's first q row

  // Q^T fragments (q pre-scaled by 0.125 in rope): [qtile j][kfrag]
  v16h qf[2][2];
#pragma unroll
  for (int j = 0; j < 2; ++j)
#pragma unroll
    for (int kf = 0; kf < 2; ++kf)
      qf[j][kf] = load_frag_b(qbase, TD, m0 + j * 16, kf * 32, lane);

  v8f z = {};
  v8f ot[4][2];                          // O^T accumulators [dtile][qtile]
#pragma unroll
  for (int dt = 0; dt < 4; ++dt)
#pragma unroll
    for (int j = 0; j < 2; ++j) ot[dt][j] = z;
  float mx[2] = {-1e30f, -1e30f};        // per-lane row stats (row = m0+j*16+l16)
  float ls[2] = {0.f, 0.f};

  int diag = m0 >> 5;                    // wave's diagonal kv tile

  for (int kt = 0; kt <= diag; ++kt) {
    int kv0 = kt * 32;

    // S^T = K * Q^T  (2 kv-subtiles i x 2 q-subtiles j, K-dim 64 = 2 WMMAs)
    v8f st[2][2];
#pragma unroll
    for (int i = 0; i < 2; ++i)
#pragma unroll
      for (int j = 0; j < 2; ++j) st[i][j] = z;
#pragma unroll
    for (int kf = 0; kf < 2; ++kf) {
      v16h k0 = load_frag_a(kbase, TD, kv0 + 0,  kf * 32, lane);
      v16h k1 = load_frag_a(kbase, TD, kv0 + 16, kf * 32, lane);
#pragma unroll
      for (int j = 0; j < 2; ++j) {
        st[0][j] = wmma_f16(k0, qf[j][kf], st[0][j]);
        st[1][j] = wmma_f16(k1, qf[j][kf], st[1][j]);
      }
    }
    if (kt == diag) {  // causal mask: only the diagonal tile needs it
#pragma unroll
      for (int i = 0; i < 2; ++i)
#pragma unroll
        for (int j = 0; j < 2; ++j) {
          int q = m0 + j * 16 + l16;
#pragma unroll
          for (int r = 0; r < 8; ++r) {
            int kv = kv0 + i * 16 + r + 8 * hi;
            if (kv > q) st[i][j][r] = -1e30f;
          }
        }
    }

    // online softmax; row n = m0 + j*16 + l16 lives in lanes {l16, l16+16}
#pragma unroll
    for (int j = 0; j < 2; ++j) {
      float tm = st[0][j][0];
#pragma unroll
      for (int r = 1; r < 8; ++r) tm = fmaxf(tm, st[0][j][r]);
#pragma unroll
      for (int r = 0; r < 8; ++r) tm = fmaxf(tm, st[1][j][r]);
      tm = fmaxf(tm, __shfl_xor(tm, 16, 32));
      float nm = fmaxf(mx[j], tm);
      float alpha = __expf(mx[j] - nm);
      mx[j] = nm;
      float rsum = 0.f;
#pragma unroll
      for (int i = 0; i < 2; ++i)
#pragma unroll
        for (int r = 0; r < 8; ++r) {
          float p = __expf(st[i][j][r] - nm);
          st[i][j][r] = p;
          rsum += p;
        }
      rsum += __shfl_xor(rsum, 16, 32);
      ls[j] = ls[j] * alpha + rsum;
#pragma unroll
      for (int dt = 0; dt < 4; ++dt)
#pragma unroll
        for (int r = 0; r < 8; ++r) ot[dt][j][r] *= alpha;
    }

    // assemble P^T B-fragments in-register: lane needs kv chunk 16*hi+0..15
    // for its q-column; the missing 8 kv values sit in the xor-16 partner.
    v16h pf[2];
#pragma unroll
    for (int j = 0; j < 2; ++j) {
      unsigned pk0[4], pk1[4], rcv[4];
#pragma unroll
      for (int t = 0; t < 4; ++t) {
        pk0[t] = pack2h(st[0][j][2 * t], st[0][j][2 * t + 1]);
        pk1[t] = pack2h(st[1][j][2 * t], st[1][j][2 * t + 1]);
      }
#pragma unroll
      for (int t = 0; t < 4; ++t) {
        unsigned snd = hi ? pk0[t] : pk1[t];
        rcv[t] = __shfl_xor(snd, 16, 32);
      }
      union { v16h v; unsigned u[8]; } f;
#pragma unroll
      for (int t = 0; t < 4; ++t) {
        f.u[t]     = hi ? rcv[t] : pk0[t];   // K = 16*hi + 0..7
        f.u[4 + t] = hi ? pk1[t] : rcv[t];   // K = 16*hi + 8..15
      }
      pf[j] = f.v;
    }

    // O^T += V^T * P^T
#pragma unroll
    for (int dt = 0; dt < 4; ++dt) {
      v16h vf = load_frag_a(vtb, S, dt * 16, kv0, lane);
      ot[dt][0] = wmma_f16(vf, pf[0], ot[dt][0]);
      ot[dt][1] = wmma_f16(vf, pf[1], ot[dt][1]);
    }
  }

  // epilogue: per lane, q-row is fixed per j and d-columns are consecutive,
  // so each (dt, j) tile stores one 16-byte vector.
#pragma unroll
  for (int j = 0; j < 2; ++j) {
    float rl = 1.0f / ls[j];
    int row = m0 + j * 16 + l16;
    _Float16* dst = att + (size_t)(b * S + row) * 1024 + h * 64 + hi * 8;
#pragma unroll
    for (int dt = 0; dt < 4; ++dt) {
      union { v8h v; _Float16 e[8]; } s;
#pragma unroll
      for (int r = 0; r < 8; ++r) s.e[r] = (_Float16)(ot[dt][j][r] * rl);
      *(v8h*)(dst + dt * 16) = s.v;
    }
  }
}

// ---------------------------------------------------------------------------
extern "C" void kernel_launch(void* const* d_in, const int* in_sizes, int n_in,
                              void* d_out, int out_size, void* d_ws, size_t ws_size,
                              hipStream_t stream) {
  const int B = 2, S = 2048, D = 1024;
  const int M = B * S;            // 4096
  const float* x      = (const float*)d_in[0];
  const int*   pos    = (const int*)d_in[1];
  const float* W_qkv  = (const float*)d_in[2];
  const float* W_o    = (const float*)d_in[3];

  char* ws = (char*)d_ws;
  _Float16* xh    = (_Float16*)(ws);                        //  8 MB
  _Float16* wqkvh = (_Float16*)(ws + (size_t)8  * 1048576); //  6 MB
  _Float16* woh   = (_Float16*)(ws + (size_t)14 * 1048576); //  2 MB
  _Float16* qkvh  = (_Float16*)(ws + (size_t)16 * 1048576); // 24 MB
  _Float16* atth  = (_Float16*)(ws + (size_t)40 * 1048576); //  8 MB
  _Float16* vth   = (_Float16*)(ws + (size_t)48 * 1048576); //  8 MB

  // 1) convert inputs to f16
  {
    int n1 = M * D;            // 4,194,304
    int n2 = 3 * D * D;        // 3,145,728
    int n3 = D * D;            // 1,048,576
    f32_to_f16_kernel<<<(n1 + 255) / 256, 256, 0, stream>>>(x, xh, n1);
    f32_to_f16_kernel<<<(n2 + 255) / 256, 256, 0, stream>>>(W_qkv, wqkvh, n2);
    f32_to_f16_kernel<<<(n3 + 255) / 256, 256, 0, stream>>>(W_o, woh, n3);
  }

  // 2) qkv = x @ W_qkv^T   (4096 x 3072 x 1024), f16 out
  gemm_wmma_kernel<1024, true><<<dim3(M / 128, 3 * D / 128), 128, 0, stream>>>(
      xh, wqkvh, (void*)qkvh, 3 * D);

  // 3) RoPE on q, k (q pre-scaled by 1/sqrt(d_k))
  rope_kernel<<<(B * S * 512) / 256, 256, 0, stream>>>(qkvh, pos);

  // 3b) one-time V transpose -> vth[(b,h)][d][s]
  transpose_v_kernel<<<dim3(S / 64, B * 16, 4), 256, 0, stream>>>(qkvh, vth);

  // 4) flash attention -> atth [B, S, D] f16
  attn_kernel<<<B * 16 * (S / 128), 128, 0, stream>>>(qkvh, vth, atth);

  // 5) out = att @ W_o^T   (4096 x 1024 x 1024), f32 out
  gemm_wmma_kernel<1024, false><<<dim3(M / 128, D / 128), 128, 0, stream>>>(
      atth, woh, d_out, D);
}
